// G2SD_18665927868990
// MI455X (gfx1250) — matrix operand order repaired and checked by reading, not compiled, not run on hardware
//
#include <hip/hip_runtime.h>
#include <hip/hip_bf16.h>
#include <math.h>

typedef __attribute__((ext_vector_type(16))) _Float16 v16h;
typedef __attribute__((ext_vector_type(8)))  _Float16 v8h;
typedef __attribute__((ext_vector_type(8)))  float    v8f;
typedef __attribute__((ext_vector_type(4)))  float    f32x4;

#define NPTS 2048
#define BATCH 8
#define MP1 16384            /* B*N, multiple of 128 */
#define GPTS 2025
#define MFOLD 16200          /* B*G */
#define MP2 16256            /* ceil(16200/128)*128 */

// ---------------------------------------------------------------------------
// Weight prep: Wt[n][k] = W[k][n]*g[n] (f16, zero padded), bias[n]=b*g+beta
// ---------------------------------------------------------------------------
__global__ __launch_bounds__(256) void prep_kernel(
    const float* __restrict__ W, const float* __restrict__ b,
    const float* __restrict__ g, const float* __restrict__ be,
    int cin, int cout, int Kp, int Np,
    _Float16* __restrict__ Wt, float* __restrict__ bias)
{
    int t = blockIdx.x * 256 + threadIdx.x;
    if (t < Np * Kp) {
        int n = t / Kp, k = t - n * Kp;
        float v = (n < cout && k < cin) ? W[k * cout + n] * g[n] : 0.0f;
        Wt[t] = (_Float16)v;
    }
    if (t < Np) bias[t] = (t < cout) ? (b[t] * g[t] + be[t]) : 0.0f;
}

// ---------------------------------------------------------------------------
// KNN: per point, top-17 nearest (self included) by squared distance
// ---------------------------------------------------------------------------
__global__ __launch_bounds__(256) void knn_kernel(
    const float* __restrict__ pts, int* __restrict__ idx)
{
    __shared__ float sx[NPTS], sy[NPTS], sz[NPTS];
    int b = blockIdx.y;
    const float* p = pts + (size_t)b * NPTS * 3;
    for (int i = threadIdx.x; i < NPTS; i += 256) {
        sx[i] = p[3 * i]; sy[i] = p[3 * i + 1]; sz[i] = p[3 * i + 2];
    }
    __syncthreads();
    int i = blockIdx.x * 256 + threadIdx.x;
    float xi = sx[i], yi = sy[i], zi = sz[i];
    float bd[17]; int bi[17];
    #pragma unroll
    for (int t = 0; t < 17; t++) { bd[t] = 1e30f; bi[t] = 0; }
    for (int j = 0; j < NPTS; j++) {
        float dx = sx[j] - xi, dy = sy[j] - yi, dz = sz[j] - zi;
        float d = dx * dx + dy * dy + dz * dz;
        #pragma unroll
        for (int t = 16; t >= 0; t--) {
            if (d < bd[t]) {
                bool shift = (t > 0) && (d < bd[t - 1]);
                if (shift) { bd[t] = bd[t - 1]; bi[t] = bi[t - 1]; }
                else       { bd[t] = d;        bi[t] = j;        }
            }
        }
    }
    int* op = idx + ((size_t)b * NPTS + i) * 17;
    #pragma unroll
    for (int t = 0; t < 17; t++) op[t] = bi[t];
}

// ---------------------------------------------------------------------------
// Neighbor aggregation MLP (scalar): 7->16->32 per neighbor, maxpool(16),
// lifted 3->32, concat 64 -> 32.  f1 written to X352 cols [0,32)
// ---------------------------------------------------------------------------
__global__ __launch_bounds__(256) void nbr_kernel(
    const float* __restrict__ pts, const int* __restrict__ idx,
    const float* W1, const float* b1, const float* g1, const float* a1,
    const float* W2, const float* b2, const float* g2, const float* a2,
    const float* Ws, const float* bs, const float* gs, const float* as,
    const float* W3, const float* b3, const float* g3, const float* a3,
    _Float16* __restrict__ out352)
{
    int r = blockIdx.x * 256 + threadIdx.x;    // < MP1
    int b = r >> 11;
    const float* base = pts + (size_t)b * NPTS * 3;
    const int* ip = idx + (size_t)r * 17;
    int i0 = ip[0];
    float ax = base[3 * i0], ay = base[3 * i0 + 1], az = base[3 * i0 + 2];
    float pooled[32];
    #pragma unroll
    for (int c = 0; c < 32; c++) pooled[c] = 0.0f;
    for (int kk = 1; kk < 17; kk++) {
        int j = ip[kk];
        float rx = base[3 * j] - ax, ry = base[3 * j + 1] - ay, rz = base[3 * j + 2] - az;
        float d = sqrtf(rx * rx + ry * ry + rz * rz + 1e-8f);
        float f[7] = { ax, ay, az, rx, ry, rz, d };
        float h1[16];
        #pragma unroll
        for (int c = 0; c < 16; c++) {
            float s = 0.0f;
            #pragma unroll
            for (int k = 0; k < 7; k++) s += f[k] * W1[k * 16 + c];
            s = (s + b1[c]) * g1[c] + a1[c];
            h1[c] = fmaxf(s, 0.0f);
        }
        #pragma unroll
        for (int c = 0; c < 32; c++) {
            float s = 0.0f;
            #pragma unroll
            for (int k = 0; k < 16; k++) s += h1[k] * W2[k * 32 + c];
            s = (s + b2[c]) * g2[c] + a2[c];
            pooled[c] = fmaxf(pooled[c], fmaxf(s, 0.0f));
        }
    }
    int n = r & (NPTS - 1);
    float px = base[3 * n], py = base[3 * n + 1], pz = base[3 * n + 2];
    float lif[32];
    #pragma unroll
    for (int c = 0; c < 32; c++) {
        float s = px * Ws[c] + py * Ws[32 + c] + pz * Ws[64 + c];
        s = (s + bs[c]) * gs[c] + as[c];
        lif[c] = fmaxf(s, 0.0f);
    }
    #pragma unroll
    for (int c = 0; c < 32; c++) {
        float s = 0.0f;
        #pragma unroll
        for (int k = 0; k < 32; k++) s += lif[k] * W3[k * 32 + c];
        #pragma unroll
        for (int k = 0; k < 32; k++) s += pooled[k] * W3[(32 + k) * 32 + c];
        s = (s + b3[c]) * g3[c] + a3[c];
        out352[(size_t)r * 352 + c] = (_Float16)fmaxf(s, 0.0f);
    }
}

// ---------------------------------------------------------------------------
// Tiled WMMA GEMM: C = act(A(MxK,f16) * Wt^T(N-major,f16) + bias [+ Res])
// Block tile 128x64, 8 waves (4 along M x 2 along N), each wave 32x32 =
// 2x2 v_wmma_f32_16x16x32_f16.  A staged via LDS in the exact CDNA5
// A-fragment layout; B fragments straight from L2-resident weights.
// ---------------------------------------------------------------------------
__global__ __launch_bounds__(256) void wmma_gemm(
    const _Float16* __restrict__ A, int lda, int M,
    const _Float16* __restrict__ Wt, int ldw,
    const float* __restrict__ bias, int Ksteps, int N,
    _Float16* __restrict__ C16, int ldc16, int coff,
    float* __restrict__ C32, int ldc32,
    const float* __restrict__ Res, int ldr, int relu)
{
    __shared__ _Float16 As[128][40];                 // 10 KB, padded rows
    int tid  = threadIdx.x;
    int lane = tid & 31;
    int wave = tid >> 5;
    int wm   = wave & 3;                             // 4 waves along M
    int wn   = wave >> 2;                            // 2 waves along N
    int ln15 = lane & 15;
    bool hi  = lane >= 16;
    int m_blk = blockIdx.y * 128;
    int n0    = blockIdx.x * 64 + wn * 32;           // this wave's 32 columns

    v8f acc[2][2] = {};

    int arow  = tid & 127;                           // A-stage loader mapping
    int ahalf = (tid >> 7) << 4;                     // 0 or 16
    const _Float16* Aptr = A + (size_t)(m_blk + arow) * lda + ahalf;

    for (int k0 = 0; k0 < Ksteps; k0 += 32) {
        // stage 128x32 A tile
        const f32x4* gp = (const f32x4*)(Aptr + k0);
        f32x4 a0 = gp[0], a1 = gp[1];
        *(f32x4*)&As[arow][ahalf]     = a0;
        *(f32x4*)&As[arow][ahalf + 8] = a1;
        if (k0 + 32 < Ksteps) __builtin_prefetch(Aptr + k0 + 32, 0, 0);
        __syncthreads();

        // A fragments (16x32): lane<16 -> K{0..7,16..23}, lane>=16 -> K{8..15,24..31}
        v16h afrag[2];
        #pragma unroll
        for (int mi = 0; mi < 2; mi++) {
            int r  = wm * 32 + mi * 16 + ln15;
            int kk = hi ? 8 : 0;
            v8h lo  = *(const v8h*)&As[r][kk];
            v8h hi8 = *(const v8h*)&As[r][kk + 16];
            afrag[mi] = __builtin_shufflevector(lo, hi8,
                0,1,2,3,4,5,6,7,8,9,10,11,12,13,14,15);
        }
        // B fragments (32x16): lane<16 -> K{0..7,16..23} of column lane, lane>=16 -> K{8..15,24..31}
        v16h bfrag[2];
        #pragma unroll
        for (int ni = 0; ni < 2; ni++) {
            const _Float16* wp = Wt + (size_t)(n0 + ni * 16 + ln15) * ldw
                                    + k0 + (hi ? 8 : 0);
            v8h b0 = *(const v8h*)wp;
            v8h b1 = *(const v8h*)(wp + 16);
            bfrag[ni] = __builtin_shufflevector(b0, b1,
                0,1,2,3,4,5,6,7,8,9,10,11,12,13,14,15);
        }
        #pragma unroll
        for (int mi = 0; mi < 2; mi++)
            #pragma unroll
            for (int ni = 0; ni < 2; ni++)
                acc[mi][ni] = __builtin_amdgcn_wmma_f32_16x16x32_f16(
                    false, afrag[mi], false, bfrag[ni],
                    (short)0, acc[mi][ni], false, false);
        __syncthreads();
    }

    // epilogue: C/D layout -> lane<16: row v, col lane ; lane>=16: row v+8
    #pragma unroll
    for (int mi = 0; mi < 2; mi++) {
        #pragma unroll
        for (int ni = 0; ni < 2; ni++) {
            int col = n0 + ni * 16 + ln15;
            if (col >= N) continue;
            float bv = bias[col];
            #pragma unroll
            for (int v = 0; v < 8; v++) {
                int row = m_blk + wm * 32 + mi * 16 + v + (hi ? 8 : 0);
                if (row >= M) continue;
                float x = acc[mi][ni][v] + bv;
                if (Res)  x += Res[(size_t)row * ldr + col];
                if (relu) x = fmaxf(x, 0.0f);
                if (C16)  C16[(size_t)row * ldc16 + coff + col] = (_Float16)x;
                if (C32)  C32[(size_t)row * ldc32 + col] = x;
            }
        }
    }
}

// ---------------------------------------------------------------------------
// Column max over N rows of one batch: out[b*ncols+c] = max_r X[b,r,coff+c]
// ---------------------------------------------------------------------------
__global__ __launch_bounds__(256) void colmax_kernel(
    const _Float16* __restrict__ X, int ld, int coff, int ncols, int rowsPerB,
    float* __restrict__ out)
{
    int c = blockIdx.x, b = blockIdx.y;
    const _Float16* p = X + (size_t)b * rowsPerB * ld + coff + c;
    float m = -1e30f;
    for (int r = threadIdx.x; r < rowsPerB; r += 256)
        m = fmaxf(m, (float)p[(size_t)r * ld]);
    __shared__ float red[256];
    red[threadIdx.x] = m; __syncthreads();
    for (int s = 128; s > 0; s >>= 1) {
        if (threadIdx.x < s) red[threadIdx.x] = fmaxf(red[threadIdx.x], red[threadIdx.x + s]);
        __syncthreads();
    }
    if (threadIdx.x == 0) out[b * ncols + c] = red[0];
}

// X128[:, 0:64] = f2 (X352 cols 32..95) ; X128[:, 64:128] = broadcast f2max
__global__ __launch_bounds__(256) void fill_x128_kernel(
    const _Float16* __restrict__ X352, const float* __restrict__ cmax,
    _Float16* __restrict__ X128)
{
    int t = blockIdx.x * 256 + threadIdx.x;      // MP1*64
    if (t >= MP1 * 64) return;
    int r = t >> 6, c = t & 63, b = r >> 11;
    X128[(size_t)r * 128 + c]      = X352[(size_t)r * 352 + 32 + c];
    X128[(size_t)r * 128 + 64 + c] = (_Float16)cmax[b * 64 + c];
}

// X352 cols 224..351 = broadcast f3max
__global__ __launch_bounds__(256) void fill_f3max_kernel(
    const float* __restrict__ cmax, _Float16* __restrict__ X352)
{
    int t = blockIdx.x * 256 + threadIdx.x;      // MP1*128
    if (t >= MP1 * 128) return;
    int r = t >> 7, c = t & 127, b = r >> 11;
    X352[(size_t)r * 352 + 224 + c] = (_Float16)cmax[b * 128 + c];
}

// softmax over N per (b,c) of scores S, weighted sum with f4 -> att[b,c]
__global__ __launch_bounds__(256) void att_reduce_kernel(
    const _Float16* __restrict__ S, const _Float16* __restrict__ F4,
    int ld, float* __restrict__ att)
{
    int c = blockIdx.x, b = blockIdx.y;
    const _Float16* sp = S  + (size_t)b * NPTS * ld + c;
    const _Float16* fp = F4 + (size_t)b * NPTS * ld + c;
    __shared__ float red[256];
    float m = -1e30f;
    for (int n = threadIdx.x; n < NPTS; n += 256)
        m = fmaxf(m, (float)sp[(size_t)n * ld]);
    red[threadIdx.x] = m; __syncthreads();
    for (int s = 128; s > 0; s >>= 1) {
        if (threadIdx.x < s) red[threadIdx.x] = fmaxf(red[threadIdx.x], red[threadIdx.x + s]);
        __syncthreads();
    }
    m = red[0]; __syncthreads();
    float se = 0.0f, wsum = 0.0f;
    for (int n = threadIdx.x; n < NPTS; n += 256) {
        float e = __expf((float)sp[(size_t)n * ld] - m);
        se += e; wsum += e * (float)fp[(size_t)n * ld];
    }
    red[threadIdx.x] = se; __syncthreads();
    for (int s = 128; s > 0; s >>= 1) {
        if (threadIdx.x < s) red[threadIdx.x] += red[threadIdx.x + s];
        __syncthreads();
    }
    se = red[0]; __syncthreads();
    red[threadIdx.x] = wsum; __syncthreads();
    for (int s = 128; s > 0; s >>= 1) {
        if (threadIdx.x < s) red[threadIdx.x] += red[threadIdx.x + s];
        __syncthreads();
    }
    if (threadIdx.x == 0) att[b * 512 + c] = red[0] / se;
}

// X_glob (128x1024 f16): row b<8 -> [f4max | att], pad rows zero
__global__ __launch_bounds__(256) void fill_glob_kernel(
    const float* __restrict__ f4max, const float* __restrict__ att,
    _Float16* __restrict__ Xg)
{
    int t = blockIdx.x * 256 + threadIdx.x;      // 128*1024
    if (t >= 128 * 1024) return;
    int row = t >> 10, c = t & 1023;
    float v = 0.0f;
    if (row < BATCH) v = (c < 512) ? f4max[row * 512 + c] : att[row * 512 + (c - 512)];
    Xg[t] = (_Float16)v;
}

// X1 (MP2 x 1056 f16): [cdw(1024) | lattice(2) | zeros]
__global__ __launch_bounds__(256) void fill_x1_kernel(
    const _Float16* __restrict__ CDW, const float* __restrict__ lat,
    _Float16* __restrict__ X1)
{
    long long t = (long long)blockIdx.x * 256 + threadIdx.x;
    if (t >= (long long)MP2 * 1056) return;
    int r = (int)(t / 1056), c = (int)(t - (long long)r * 1056);
    _Float16 v = (_Float16)0.0f;
    if (r < MFOLD) {
        int b = r / GPTS, g = r - b * GPTS;
        if (c < 1024)       v = CDW[b * 1024 + c];
        else if (c == 1024) v = (_Float16)lat[2 * g];
        else if (c == 1025) v = (_Float16)lat[2 * g + 1];
    }
    X1[t] = v;
}

// overwrite X1 cols 1024..1026 with rec1 (3 channels)
__global__ __launch_bounds__(256) void fill_x2_kernel(
    const _Float16* __restrict__ REC1, _Float16* __restrict__ X1)
{
    int r = blockIdx.x * 256 + threadIdx.x;
    if (r >= MP2) return;
    _Float16 a = (_Float16)0.0f, b = a, c = a;
    if (r < MFOLD) { a = REC1[r * 4]; b = REC1[r * 4 + 1]; c = REC1[r * 4 + 2]; }
    X1[(size_t)r * 1056 + 1024] = a;
    X1[(size_t)r * 1056 + 1025] = b;
    X1[(size_t)r * 1056 + 1026] = c;
}

// ---------------------------------------------------------------------------
// Host side
// ---------------------------------------------------------------------------
static const int L_CIN[23]  = { 7,16,3,64, 32,64,32, 128,128,128, 352, 512,
                                1024,512,1024, 1026,256,128,64, 1027,256,128,64 };
static const int L_COUT[23] = { 16,32,32,32, 64,64,64, 128,128,128, 512, 512,
                                512,1024,1024, 256,128,64,3, 256,128,64,3 };

static inline int rup(int x, int a) { return (x + a - 1) / a * a; }

static void launch_gemm(hipStream_t s,
                        const _Float16* A, int lda, int M,
                        const _Float16* Wt, int ldw, const float* bias,
                        int Ksteps, int N,
                        _Float16* C16, int ldc16, int coff,
                        float* C32, int ldc32,
                        const float* Res, int ldr, int relu)
{
    dim3 grid((N + 63) / 64, (M + 127) / 128);
    wmma_gemm<<<grid, 256, 0, s>>>(A, lda, M, Wt, ldw, bias, Ksteps, N,
                                   C16, ldc16, coff, C32, ldc32, Res, ldr, relu);
}

extern "C" void kernel_launch(void* const* d_in, const int* in_sizes, int n_in,
                              void* d_out, int out_size, void* d_ws, size_t ws_size,
                              hipStream_t stream)
{
    (void)in_sizes; (void)n_in; (void)out_size; (void)ws_size;
    const float* pts = (const float*)d_in[0];
    const float* lat = (const float*)d_in[1 + 4 * 23];      // lattice
    float* out = (float*)d_out;
    unsigned char* ws = (unsigned char*)d_ws;

    // ---- workspace layout (bump allocator, 256B aligned) ----
    size_t off = 0;
    auto bump = [&](size_t bytes) -> size_t {
        size_t o = (off + 255) & ~(size_t)255; off = o + bytes; return o;
    };
    size_t wtOff[23], bOff[23];
    for (int l = 0; l < 23; l++) {
        int Kp = rup(L_CIN[l], 32), Np = rup(L_COUT[l], 64);
        wtOff[l] = bump((size_t)Np * Kp * 2);
        bOff[l]  = bump((size_t)Np * 4);
    }
    size_t cmaxOff = bump(8 * 512 * 4);
    size_t attOff  = bump(8 * 512 * 4);
    size_t globOff = bump(128 * 1024 * 2);
    size_t fc1Off  = bump(128 * 512 * 2);
    size_t fc2Off  = bump(128 * 1024 * 2);
    size_t cdwOff  = bump(128 * 1024 * 2);

    // union region: backbone scratch and fold scratch never overlap in time
    size_t u = 0;
    auto ub = [&](size_t bytes) -> size_t {
        size_t o = (u + 255) & ~(size_t)255; u = o + bytes; return o;
    };
    size_t idxU = ub((size_t)MP1 * 17 * 4);
    size_t x352U = ub((size_t)MP1 * 352 * 2);
    size_t x128U = ub((size_t)MP1 * 128 * 2);
    size_t t1U   = ub((size_t)MP1 * 128 * 2);
    size_t t2U   = ub((size_t)MP1 * 128 * 4);
    size_t x4U   = ub((size_t)MP1 * 512 * 2);
    size_t sU    = ub((size_t)MP1 * 512 * 2);
    size_t backBytes = u;
    u = 0;
    size_t x1U   = ub((size_t)MP2 * 1056 * 2);
    size_t y1U   = ub((size_t)MP2 * 256 * 2);
    size_t y2U   = ub((size_t)MP2 * 128 * 2);
    size_t y3U   = ub((size_t)MP2 * 64 * 2);
    size_t rec1U = ub((size_t)MP2 * 4 * 2);
    size_t foldBytes = u;
    size_t uniOff = bump(backBytes > foldBytes ? backBytes : foldBytes);

    _Float16* Wt[23]; float* Bs[23];
    for (int l = 0; l < 23; l++) {
        Wt[l] = (_Float16*)(ws + wtOff[l]);
        Bs[l] = (float*)(ws + bOff[l]);
    }
    float*     cmax  = (float*)(ws + cmaxOff);
    float*     attB  = (float*)(ws + attOff);
    _Float16*  Xg    = (_Float16*)(ws + globOff);
    _Float16*  Xfc1  = (_Float16*)(ws + fc1Off);
    _Float16*  Xfc2  = (_Float16*)(ws + fc2Off);
    _Float16*  CDW   = (_Float16*)(ws + cdwOff);
    int*       idx   = (int*)(ws + uniOff + idxU);
    _Float16*  X352  = (_Float16*)(ws + uniOff + x352U);
    _Float16*  X128  = (_Float16*)(ws + uniOff + x128U);
    _Float16*  T1    = (_Float16*)(ws + uniOff + t1U);
    float*     T2    = (float*)(ws + uniOff + t2U);
    _Float16*  X4    = (_Float16*)(ws + uniOff + x4U);
    _Float16*  Sc    = (_Float16*)(ws + uniOff + sU);
    _Float16*  X1    = (_Float16*)(ws + uniOff + x1U);
    _Float16*  Y1    = (_Float16*)(ws + uniOff + y1U);
    _Float16*  Y2    = (_Float16*)(ws + uniOff + y2U);
    _Float16*  Y3    = (_Float16*)(ws + uniOff + y3U);
    _Float16*  REC1  = (_Float16*)(ws + uniOff + rec1U);

    // ---- 1) weight prep (fold BN, transpose to N-major f16) ----
    for (int l = 0; l < 23; l++) {
        const float* W  = (const float*)d_in[1 + 4 * l];
        const float* bb = (const float*)d_in[2 + 4 * l];
        const float* gg = (const float*)d_in[3 + 4 * l];
        const float* be = (const float*)d_in[4 + 4 * l];
        int Kp = rup(L_CIN[l], 32), Np = rup(L_COUT[l], 64);
        int tot = Np * Kp;
        prep_kernel<<<(tot + 255) / 256, 256, 0, stream>>>(
            W, bb, gg, be, L_CIN[l], L_COUT[l], Kp, Np, Wt[l], Bs[l]);
    }

    // ---- 2) KNN + neighbor MLP -> f1 (X352 cols 0..31) ----
    knn_kernel<<<dim3(NPTS / 256, BATCH), 256, 0, stream>>>(pts, idx);
    nbr_kernel<<<MP1 / 256, 256, 0, stream>>>(
        pts, idx,
        (const float*)d_in[1],  (const float*)d_in[2],  (const float*)d_in[3],  (const float*)d_in[4],
        (const float*)d_in[5],  (const float*)d_in[6],  (const float*)d_in[7],  (const float*)d_in[8],
        (const float*)d_in[9],  (const float*)d_in[10], (const float*)d_in[11], (const float*)d_in[12],
        (const float*)d_in[13], (const float*)d_in[14], (const float*)d_in[15], (const float*)d_in[16],
        X352);

    // ---- 3) res1: f2 = relu(m2(relu(m1(f1))) + sc(f1)) -> X352 cols 32..95 ----
    launch_gemm(stream, X352, 352, MP1, Wt[4], 32, Bs[4], 32, 64,
                T1, 64, 0, nullptr, 0, nullptr, 0, 1);
    launch_gemm(stream, T1, 64, MP1, Wt[5], 64, Bs[5], 64, 64,
                nullptr, 0, 0, T2, 64, nullptr, 0, 0);
    launch_gemm(stream, X352, 352, MP1, Wt[6], 32, Bs[6], 32, 64,
                X352, 352, 32, nullptr, 0, T2, 64, 1);

    // f2max + concat -> X128
    colmax_kernel<<<dim3(64, BATCH), 256, 0, stream>>>(X352, 352, 32, 64, NPTS, cmax);
    fill_x128_kernel<<<(MP1 * 64) / 256, 256, 0, stream>>>(X352, cmax, X128);

    // ---- 4) res2 -> f3 (X352 cols 96..223) ----
    launch_gemm(stream, X128, 128, MP1, Wt[7], 128, Bs[7], 128, 128,
                T1, 128, 0, nullptr, 0, nullptr, 0, 1);
    launch_gemm(stream, T1, 128, MP1, Wt[8], 128, Bs[8], 128, 128,
                nullptr, 0, 0, T2, 128, nullptr, 0, 0);
    launch_gemm(stream, X128, 128, MP1, Wt[9], 128, Bs[9], 128, 128,
                X352, 352, 96, nullptr, 0, T2, 128, 1);

    // f3max broadcast -> X352 cols 224..351
    colmax_kernel<<<dim3(128, BATCH), 256, 0, stream>>>(X352, 352, 96, 128, NPTS, cmax);
    fill_f3max_kernel<<<(MP1 * 128) / 256, 256, 0, stream>>>(cmax, X352);

    // ---- 5) fuse -> f4 (X4), att scores (Sc), attention pooling ----
    launch_gemm(stream, X352, 352, MP1, Wt[10], 352, Bs[10], 352, 512,
                X4, 512, 0, nullptr, 0, nullptr, 0, 1);
    launch_gemm(stream, X4, 512, MP1, Wt[11], 512, Bs[11], 512, 512,
                Sc, 512, 0, nullptr, 0, nullptr, 0, 0);
    colmax_kernel<<<dim3(512, BATCH), 256, 0, stream>>>(X4, 512, 0, 512, NPTS, cmax);
    att_reduce_kernel<<<dim3(512, BATCH), 256, 0, stream>>>(Sc, X4, 512, attB);
    fill_glob_kernel<<<(128 * 1024) / 256, 256, 0, stream>>>(cmax, attB, Xg);

    // ---- 6) FC head -> codeword ----
    launch_gemm(stream, Xg, 1024, BATCH, Wt[12], 1024, Bs[12], 1024, 512,
                Xfc1, 512, 0, nullptr, 0, nullptr, 0, 1);
    launch_gemm(stream, Xfc1, 512, BATCH, Wt[13], 512, Bs[13], 512, 1024,
                Xfc2, 1024, 0, nullptr, 0, nullptr, 0, 1);
    launch_gemm(stream, Xfc2, 1024, BATCH, Wt[14], 1024, Bs[14], 1024, 1024,
                CDW, 1024, 0, nullptr, 0, nullptr, 0, 0);

    // ---- 7) fold1 ----
    long long x1tot = (long long)MP2 * 1056;
    fill_x1_kernel<<<(unsigned)((x1tot + 255) / 256), 256, 0, stream>>>(CDW, lat, X1);
    launch_gemm(stream, X1, 1056, MFOLD, Wt[15], 1056, Bs[15], 1056, 256,
                Y1, 256, 0, nullptr, 0, nullptr, 0, 1);
    launch_gemm(stream, Y1, 256, MFOLD, Wt[16], 256, Bs[16], 256, 128,
                Y2, 128, 0, nullptr, 0, nullptr, 0, 1);
    launch_gemm(stream, Y2, 128, MFOLD, Wt[17], 128, Bs[17], 128, 64,
                Y3, 64, 0, nullptr, 0, nullptr, 0, 1);
    launch_gemm(stream, Y3, 64, MFOLD, Wt[18], 64, Bs[18], 64, 3,
                REC1, 4, 0, nullptr, 0, nullptr, 0, 0);

    // ---- 8) fold2 (reuse X1, overwrite cols 1024..1026 with rec1) ----
    fill_x2_kernel<<<(MP2 + 255) / 256, 256, 0, stream>>>(REC1, X1);
    launch_gemm(stream, X1, 1056, MFOLD, Wt[19], 1056, Bs[19], 1056, 256,
                Y1, 256, 0, nullptr, 0, nullptr, 0, 1);
    launch_gemm(stream, Y1, 256, MFOLD, Wt[20], 256, Bs[20], 256, 128,
                Y2, 128, 0, nullptr, 0, nullptr, 0, 1);
    launch_gemm(stream, Y2, 128, MFOLD, Wt[21], 128, Bs[21], 128, 64,
                Y3, 64, 0, nullptr, 0, nullptr, 0, 1);
    launch_gemm(stream, Y3, 64, MFOLD, Wt[22], 64, Bs[22], 64, 3,
                nullptr, 0, 0, out, 3, nullptr, 0, 0);
}
